// GTrXL_43241730736314
// MI455X (gfx1250) — matrix operand order repaired and checked by reading, not compiled, not run on hardware
//
#include <hip/hip_runtime.h>
#include <stdint.h>

typedef __attribute__((ext_vector_type(16))) __bf16 v16bf;
typedef __attribute__((ext_vector_type(8)))  float  v8f;
typedef unsigned long long u64;
typedef int v4i_b __attribute__((vector_size(16)));   // matches builtin param type

#define DEVFN static __device__ __forceinline__

#if defined(__gfx1250__) && __has_builtin(__builtin_amdgcn_global_load_async_to_lds_b128) && __has_builtin(__builtin_amdgcn_s_wait_asynccnt)
#define HAS_ASYNC_LDS 1
#else
#define HAS_ASYNC_LDS 0
#endif

DEVFN unsigned short f2bf(float f) {
  unsigned u = __float_as_uint(f);
  u += 0x7FFFu + ((u >> 16) & 1u);   // round-to-nearest-even
  return (unsigned short)(u >> 16);
}
DEVFN float sigm(float x) { return 1.0f / (1.0f + __expf(-x)); }

#if HAS_ASYNC_LDS
DEVFN void async_cp16(const void* g, void* l) {
  __builtin_amdgcn_global_load_async_to_lds_b128(
      (v4i_b*)g,
      (__attribute__((address_space(3))) v4i_b*)l,
      0, 0);
}
#endif

union Frag {
  v16bf v;
  u64 q[4];
  unsigned short u[16];
};

enum { GF_TRANSB = 1, GF_ACC = 2, GF_RELU = 4, GF_OUTBF16 = 8 };

// C[M,N] = A[M,K](bf16) * B(bf16), compile-time FLAGS (no in-loop branching).
// Block: 128 threads = 4 waves; block tile 128x64, K-step 32.
// Wave tile 32x64 -> 8 v_wmma_f32_16x16x32_bf16 per K-step.
// TRANSB (B stored [N][K]): async global->LDS double-buffered pipeline when available.
// LDS rows padded to 80B: 16B-aligned for b128 async writes, 20-bank skew.
// Assumes M%128==0, N%64==0, K%32==0 (true for every GEMM in this model).
template <int FLAGS>
__global__ __launch_bounds__(128) void k_gemm(
    const unsigned short* __restrict__ A,
    const unsigned short* __restrict__ B,
    void* __restrict__ C,
    const float* __restrict__ bias,
    int M, int N, int K, int lda, int ldb, int ldc,
    long long sA, long long sB, long long sC, float cconst)
{
  constexpr bool TB = (FLAGS & GF_TRANSB) != 0;
  __shared__ unsigned short As[2][128][40];
  __shared__ unsigned short Bs[2][64][40];
  const int tid  = threadIdx.x;
  const int wave = tid >> 5;
  const int lane = tid & 31;
  const int m0 = blockIdx.y * 128;
  const int n0 = blockIdx.x * 64;
  A += (long long)blockIdx.z * sA + (long long)(m0 + tid) * lda;  // per-thread A row
  B += (long long)blockIdx.z * sB;

  v8f acc[2][4];
  #pragma unroll
  for (int a = 0; a < 2; ++a)
    #pragma unroll
    for (int j = 0; j < 4; ++j)
      #pragma unroll
      for (int e = 0; e < 8; ++e) acc[a][j][e] = 0.0f;

  const int mrow  = (wave << 5) + (lane & 15);
  const int khalf = (lane & 16) ? 8 : 0;   // A frag: hi half-wave holds K+8 pattern
  const int kbb   = (lane & 16) ? 8 : 0;   // B frag: hi half-wave holds K+8 pattern

  auto compute = [&](int buf) {
    Frag a0, a1, bf[4];
    const unsigned short* p0 = &As[buf][mrow][khalf];
    a0.q[0] = *(const u64*)(p0);      a0.q[1] = *(const u64*)(p0 + 4);
    a0.q[2] = *(const u64*)(p0 + 16); a0.q[3] = *(const u64*)(p0 + 20);
    const unsigned short* p1 = &As[buf][mrow + 16][khalf];
    a1.q[0] = *(const u64*)(p1);      a1.q[1] = *(const u64*)(p1 + 4);
    a1.q[2] = *(const u64*)(p1 + 16); a1.q[3] = *(const u64*)(p1 + 20);
    #pragma unroll
    for (int j = 0; j < 4; ++j) {
      const unsigned short* pb = &Bs[buf][(j << 4) + (lane & 15)][kbb];
      bf[j].q[0] = *(const u64*)(pb);      bf[j].q[1] = *(const u64*)(pb + 4);
      bf[j].q[2] = *(const u64*)(pb + 16); bf[j].q[3] = *(const u64*)(pb + 20);
    }
    #pragma unroll
    for (int j = 0; j < 4; ++j) {
      acc[0][j] = __builtin_amdgcn_wmma_f32_16x16x32_bf16(
          false, a0.v, false, bf[j].v, (short)0, acc[0][j], false, false);
      acc[1][j] = __builtin_amdgcn_wmma_f32_16x16x32_bf16(
          false, a1.v, false, bf[j].v, (short)0, acc[1][j], false, false);
    }
  };

  const int nsteps = K >> 5;

  if constexpr (TB) {
#if HAS_ASYNC_LDS
    // ---- async global->LDS, double-buffered: copy(step s+1) overlaps wmma(step s)
    const int bn = tid >> 1, bks = (tid & 1) * 16;
    const unsigned short* Brow = B + (long long)(n0 + bn) * ldb + bks;
    auto issue = [&](int s, int buf) {
      const unsigned short* ga = A + (s << 5);
      #pragma unroll
      for (int c = 0; c < 4; ++c)
        async_cp16(ga + c * 8, &As[buf][tid][c * 8]);
      const unsigned short* gb = Brow + (s << 5);
      #pragma unroll
      for (int c = 0; c < 2; ++c)
        async_cp16(gb + c * 8, &Bs[buf][bn][bks + c * 8]);
    };
    issue(0, 0);
    for (int s = 0; s < nsteps; ++s) {
      __builtin_amdgcn_s_wait_asynccnt(0);
      __syncthreads();                       // tile(s) resident for whole block
      if (s + 1 < nsteps) issue(s + 1, (s + 1) & 1);
      compute(s & 1);
    }
#else
    // ---- sync fallback: regs -> LDS with barriers
    const int bn = tid >> 1, bks = (tid & 1) * 16;
    for (int s = 0; s < nsteps; ++s) {
      const int k0 = s << 5;
      u64 ar[8];
      const u64* ag = (const u64*)(A + k0);
      #pragma unroll
      for (int c = 0; c < 8; ++c) ar[c] = ag[c];
      u64 br[4];
      const u64* bg = (const u64*)(B + (long long)(n0 + bn) * ldb + k0 + bks);
      #pragma unroll
      for (int c = 0; c < 4; ++c) br[c] = bg[c];
      if (s + 1 < nsteps) __builtin_prefetch(A + k0 + 32, 0, 1);
      __syncthreads();
      u64* da = (u64*)&As[0][tid][0];
      #pragma unroll
      for (int c = 0; c < 8; ++c) da[c] = ar[c];
      u64* db = (u64*)&Bs[0][bn][bks];
      #pragma unroll
      for (int c = 0; c < 4; ++c) db[c] = br[c];
      __syncthreads();
      compute(0);
    }
#endif
  } else {
    // ---- B stored [K][N]: sync path with transpose into LDS (only the P.V GEMM)
    const int bk = tid >> 2, bns = (tid & 3) * 16;
    for (int s = 0; s < nsteps; ++s) {
      const int k0 = s << 5;
      u64 ar[8];
      const u64* ag = (const u64*)(A + k0);
      #pragma unroll
      for (int c = 0; c < 8; ++c) ar[c] = ag[c];
      u64 br[4];
      const u64* bg = (const u64*)(B + (long long)(k0 + bk) * ldb + n0 + bns);
      #pragma unroll
      for (int c = 0; c < 4; ++c) br[c] = bg[c];
      if (s + 1 < nsteps) __builtin_prefetch(A + k0 + 32, 0, 1);
      __syncthreads();
      u64* da = (u64*)&As[0][tid][0];
      #pragma unroll
      for (int c = 0; c < 8; ++c) da[c] = ar[c];
      union { u64 q[4]; unsigned short u[16]; } t;
      #pragma unroll
      for (int c = 0; c < 4; ++c) t.q[c] = br[c];
      #pragma unroll
      for (int e = 0; e < 16; ++e) Bs[0][bns + e][bk] = t.u[e];
      __syncthreads();
      compute(0);
    }
  }

  // ---- epilogue: C/D VGPR layout (lanes 16-31 hold M=8+i)
  const int mloc = (lane & 16) ? 8 : 0;
  const int nloc = lane & 15;
  const long long cb = (long long)blockIdx.z * sC;
  #pragma unroll
  for (int a = 0; a < 2; ++a)
    #pragma unroll
    for (int j = 0; j < 4; ++j)
      #pragma unroll
      for (int i = 0; i < 8; ++i) {
        int row = m0 + (wave << 5) + (a << 4) + mloc + i;
        int col = n0 + (j << 4) + nloc;
        long long off = cb + (long long)row * ldc + col;
        float val = acc[a][j][i];
        if constexpr ((FLAGS & GF_ACC) != 0) val += ((const float*)C)[off];
        if (bias) val += bias[col];
        val += cconst;
        if constexpr ((FLAGS & GF_RELU) != 0) val = fmaxf(val, 0.0f);
        if constexpr ((FLAGS & GF_OUTBF16) != 0) ((unsigned short*)C)[off] = f2bf(val);
        else                                     ((float*)C)[off] = val;
      }
}

__global__ void k_f2b(const float* __restrict__ s, unsigned short* __restrict__ d, long long n) {
  long long id = (long long)blockIdx.x * 256 + threadIdx.x;
  if (id < n) d[id] = f2bf(s[id]);
}

// f32 [K][N] -> bf16 [N][K] (LDS-tiled transpose); batched via blockIdx.z.
__global__ __launch_bounds__(256) void k_f2bT(const float* __restrict__ s,
                                              unsigned short* __restrict__ d,
                                              int K, int N, long long sS, long long sD) {
  __shared__ float tile[32][33];
  s += (long long)blockIdx.z * sS;
  d += (long long)blockIdx.z * sD;
  const int k0 = blockIdx.y * 32, n0 = blockIdx.x * 32;
  const int tx = threadIdx.x & 31, ty = threadIdx.x >> 5;   // 32 x 8
  #pragma unroll
  for (int r = ty; r < 32; r += 8) tile[r][tx] = s[(long long)(k0 + r) * N + n0 + tx];
  __syncthreads();
  #pragma unroll
  for (int r = ty; r < 32; r += 8)
    d[(long long)(n0 + r) * K + k0 + tx] = f2bf(tile[tx][r]);
}

__global__ void k_posemb(unsigned short* __restrict__ dst, int klen, int D) {
  long long id = (long long)blockIdx.x * 256 + threadIdx.x;
  if (id >= (long long)klen * D) return;
  int c = (int)(id % D), j = (int)(id / D);
  float pos = (float)(klen - 1 - j);
  int half = D >> 1;
  int c2 = (c < half) ? c : (c - half);
  float invf = __expf(-9.2103403719761836f * ((float)(2 * c2) / (float)D)); // 10000^(-2c/D)
  float a = pos * invf;
  dst[id] = f2bf((c < half) ? sinf(a) : cosf(a));
}

// softmax over k with Transformer-XL rel_shift gather + causal-memory mask.
// grid (q, H), block 256; assumes klen == 1024.
__global__ __launch_bounds__(256) void k_softmax_rel(
    const float* __restrict__ AC, const float* __restrict__ BD,
    unsigned short* __restrict__ P, int q, int klen, int mlen, float scale)
{
  const int i = blockIdx.x, h = blockIdx.y, tid = threadIdx.x;
  const float* ac = AC + ((long long)h * q + i) * klen;
  const float* bd = BD + (long long)h * q * klen;
  unsigned short* pr = P + ((long long)h * q + i) * klen;
  __shared__ float red[256];
  float v[4];
  float mx = -3.0e38f;
  #pragma unroll
  for (int c = 0; c < 4; ++c) {
    int j = (c << 8) + tid;
    int p = i * klen + j + q;                 // rel_shift flat index
    int ip = p / (klen + 1);
    int j2 = p - ip * (klen + 1);
    float bdv = (j2 == 0) ? 0.0f : bd[(long long)ip * klen + (j2 - 1)];
    float s = (ac[j] + bdv) * scale;
    v[c] = ((j - i) > mlen) ? -3.0e38f : s;
    mx = fmaxf(mx, v[c]);
  }
  red[tid] = mx; __syncthreads();
  for (int st = 128; st > 0; st >>= 1) { if (tid < st) red[tid] = fmaxf(red[tid], red[tid + st]); __syncthreads(); }
  mx = red[0]; __syncthreads();
  float sum = 0.f;
  #pragma unroll
  for (int c = 0; c < 4; ++c) {
    float e = (v[c] < -1.0e38f) ? 0.0f : __expf(v[c] - mx);
    v[c] = e; sum += e;
  }
  red[tid] = sum; __syncthreads();
  for (int st = 128; st > 0; st >>= 1) { if (tid < st) red[tid] += red[tid + st]; __syncthreads(); }
  float inv = 1.0f / red[0];
  #pragma unroll
  for (int c = 0; c < 4; ++c) pr[(c << 8) + tid] = f2bf(v[c] * inv);
}

__global__ __launch_bounds__(256) void k_layernorm_bf16(
    const float* __restrict__ x, unsigned short* __restrict__ out, int D)
{
  const int row = blockIdx.x, tid = threadIdx.x;
  const float* xr = x + (long long)row * D;
  __shared__ float red[256];
  float s = 0.f;
  for (int j = tid; j < D; j += 256) s += xr[j];
  red[tid] = s; __syncthreads();
  for (int st = 128; st > 0; st >>= 1) { if (tid < st) red[tid] += red[tid + st]; __syncthreads(); }
  float mu = red[0] / (float)D; __syncthreads();
  float s2 = 0.f;
  for (int j = tid; j < D; j += 256) { float dd = xr[j] - mu; s2 += dd * dd; }
  red[tid] = s2; __syncthreads();
  for (int st = 128; st > 0; st >>= 1) { if (tid < st) red[tid] += red[tid + st]; __syncthreads(); }
  float rs = rsqrtf(red[0] / (float)D + 1e-5f);
  for (int j = tid; j < D; j += 256) out[(long long)row * D + j] = f2bf((xr[j] - mu) * rs);
}

__global__ void k_gru_ew1(const float* __restrict__ t0, float* __restrict__ t1,
                          const float* __restrict__ x, unsigned short* __restrict__ rx,
                          long long n)
{
  long long id = (long long)blockIdx.x * 256 + threadIdx.x;
  if (id >= n) return;
  float r = sigm(t0[id]);
  float z = sigm(t1[id]);      // bias (+b - BG) already added by GEMM epilogue
  t1[id] = z;
  rx[id] = f2bf(r * x[id]);
}

__global__ void k_gru_ew2(const float* __restrict__ t2, const float* __restrict__ z,
                          const float* __restrict__ x, float* __restrict__ outf,
                          unsigned short* __restrict__ outb, long long n)
{
  long long id = (long long)blockIdx.x * 256 + threadIdx.x;
  if (id >= n) return;
  float h = tanhf(t2[id]);
  float zz = z[id];
  float o = (1.0f - zz) * x[id] + zz * h;
  outf[id] = o;
  outb[id] = f2bf(o);
}

// hs: [L+1][T][B][D] contiguous -> logits [T][B][(L+1)*D]
__global__ void k_scatter_logits(const float* __restrict__ hs, float* __restrict__ lg,
                                 int T, int B, int D, int L1)
{
  long long id = (long long)blockIdx.x * 256 + threadIdx.x;
  long long tot = (long long)L1 * T * B * D;
  if (id >= tot) return;
  int d = (int)(id % D); long long r = id / D;
  int b = (int)(r % B);  r /= B;
  int t = (int)(r % T);
  int l = (int)(r / T);
  lg[(((long long)t * B + b) * L1 + l) * D + d] = hs[id];
}

extern "C" void kernel_launch(void* const* d_in, const int* in_sizes, int n_in,
                              void* d_out, int out_size, void* d_ws, size_t ws_size,
                              hipStream_t stream) {
  (void)in_sizes; (void)n_in; (void)out_size; (void)ws_size;
  const int T = 512, Bb = 8, D = 1024, H = 16, DH = 64, FFd = 4096, L = 4, Mm = 512;
  const int KL = Mm + T;                    // 1024
  const long long nTBD = (long long)T * Bb * D;

  const float* in_inputs = (const float*)d_in[0];
  const float* in_memory = (const float*)d_in[1];
  const float* qkv_w     = (const float*)d_in[2];
  const float* r_net_w   = (const float*)d_in[3];
  const float* out_w     = (const float*)d_in[4];
  const float* ff_w1     = (const float*)d_in[5];
  const float* ff_b1     = (const float*)d_in[6];
  const float* ff_w2     = (const float*)d_in[7];
  const float* ff_b2     = (const float*)d_in[8];
  const float* gates_w   = (const float*)d_in[9];
  const float* gates_b   = (const float*)d_in[10];
  const float* r_w_bias  = (const float*)d_in[11];
  const float* r_r_bias  = (const float*)d_in[12];

  float* logits = (float*)d_out;                        // [T][B][(L+1)*D]
  float* hs = logits + (long long)T * Bb * (L + 1) * D; // new_memory region = hidden states

  // ---- workspace carve (bump allocator, 256B aligned)
  char* wsp = (char*)d_ws; size_t wo = 0;
  auto walloc = [&](size_t bytes) -> void* {
    void* p = wsp + wo; wo = (wo + bytes + 255) & ~(size_t)255; return p;
  };
  unsigned short* catb  = (unsigned short*)walloc((size_t)KL * Bb * D * 2);
  unsigned short* posb  = (unsigned short*)walloc((size_t)KL * D * 2);
  unsigned short* rkb   = (unsigned short*)walloc((size_t)KL * H * DH * 2);
  unsigned short* qwb   = (unsigned short*)walloc((size_t)T * Bb * D * 2);
  unsigned short* qrb   = (unsigned short*)walloc((size_t)T * Bb * D * 2);
  unsigned short* kbuf  = (unsigned short*)walloc((size_t)KL * Bb * D * 2);
  unsigned short* vbuf  = (unsigned short*)walloc((size_t)KL * Bb * D * 2);
  float*          scA   = (float*)walloc((size_t)H * T * KL * 4);
  float*          scB   = (float*)walloc((size_t)H * T * KL * 4);
  unsigned short* prb   = (unsigned short*)walloc((size_t)H * T * KL * 2);
  unsigned short* avb   = (unsigned short*)walloc((size_t)T * Bb * D * 2);
  float*          attno = (float*)walloc((size_t)T * Bb * D * 4);
  unsigned short* yb    = (unsigned short*)walloc((size_t)T * Bb * D * 2);
  float*          t0buf = (float*)walloc((size_t)T * Bb * D * 4);
  float*          t1buf = (float*)walloc((size_t)T * Bb * D * 4);
  float*          t2buf = (float*)walloc((size_t)T * Bb * D * 4);
  unsigned short* rxb   = (unsigned short*)walloc((size_t)T * Bb * D * 2);
  float*          srcf  = (float*)walloc((size_t)T * Bb * D * 4);
  unsigned short* srcb  = (unsigned short*)walloc((size_t)T * Bb * D * 2);
  unsigned short* h1b   = (unsigned short*)walloc((size_t)T * Bb * FFd * 2);
  float*          h2buf = (float*)walloc((size_t)T * Bb * D * 4);
  unsigned short* ffbuf = (unsigned short*)walloc((size_t)T * Bb * D * 2);
  // transposed bf16 weights ([N][K] layout for the async TRANSB path)
  unsigned short* wqkvT = (unsigned short*)walloc((size_t)3 * D * D * 2);
  unsigned short* wrnT  = (unsigned short*)walloc((size_t)D * D * 2);
  unsigned short* woutT = (unsigned short*)walloc((size_t)D * D * 2);
  unsigned short* wff1T = (unsigned short*)walloc((size_t)D * FFd * 2);
  unsigned short* wff2T = (unsigned short*)walloc((size_t)FFd * D * 2);
  unsigned short* wgT   = (unsigned short*)walloc((size_t)12 * D * D * 2);

  auto gemm = [&](int flags, const unsigned short* Ap, const unsigned short* Bp, void* Cp,
                  const float* biasp, int Mr, int Nc, int Kd,
                  int lda, int ldb, int ldc,
                  long long sAa, long long sBb, long long sCc, int nbatch, float cc) {
    dim3 grid((unsigned)(Nc / 64), (unsigned)(Mr / 128), (unsigned)nbatch);
    dim3 blk(128);
    switch (flags) {
      case GF_TRANSB:
        k_gemm<GF_TRANSB><<<grid, blk, 0, stream>>>(Ap, Bp, Cp, biasp, Mr, Nc, Kd, lda, ldb, ldc, sAa, sBb, sCc, cc); break;
      case (GF_TRANSB | GF_OUTBF16):
        k_gemm<(GF_TRANSB | GF_OUTBF16)><<<grid, blk, 0, stream>>>(Ap, Bp, Cp, biasp, Mr, Nc, Kd, lda, ldb, ldc, sAa, sBb, sCc, cc); break;
      case (GF_TRANSB | GF_ACC):
        k_gemm<(GF_TRANSB | GF_ACC)><<<grid, blk, 0, stream>>>(Ap, Bp, Cp, biasp, Mr, Nc, Kd, lda, ldb, ldc, sAa, sBb, sCc, cc); break;
      case (GF_TRANSB | GF_RELU | GF_OUTBF16):
        k_gemm<(GF_TRANSB | GF_RELU | GF_OUTBF16)><<<grid, blk, 0, stream>>>(Ap, Bp, Cp, biasp, Mr, Nc, Kd, lda, ldb, ldc, sAa, sBb, sCc, cc); break;
      default: // GF_OUTBF16 (non-transB P.V)
        k_gemm<GF_OUTBF16><<<grid, blk, 0, stream>>>(Ap, Bp, Cp, biasp, Mr, Nc, Kd, lda, ldb, ldc, sAa, sBb, sCc, cc); break;
    }
  };
  auto conv = [&](const float* s, unsigned short* dptr, long long n) {
    k_f2b<<<(unsigned)((n + 255) / 256), 256, 0, stream>>>(s, dptr, n);
  };
  auto convT = [&](const float* s, unsigned short* dptr, int K, int N, int nbatch) {
    dim3 grid((unsigned)(N / 32), (unsigned)(K / 32), (unsigned)nbatch);
    k_f2bT<<<grid, 256, 0, stream>>>(s, dptr, K, N, (long long)K * N, (long long)K * N);
  };

  // hidden_states[0] = inputs (m_len == q_len, so new_memory[l] == hidden_states[l])
  (void)hipMemcpyAsync(hs, in_inputs, sizeof(float) * nTBD, hipMemcpyDeviceToDevice, stream);
  k_posemb<<<(unsigned)(((long long)KL * D + 255) / 256), 256, 0, stream>>>(posb, KL, D);
  conv(in_inputs, catb + (long long)Mm * Bb * D, nTBD);   // layer_out -> cat tail (bf16)

  for (int l = 0; l < L; ++l) {
    conv(in_memory + (long long)l * Mm * Bb * D, catb, (long long)Mm * Bb * D);
    convT(qkv_w + (long long)l * D * 3 * D, wqkvT, D, 3 * D, 1);
    convT(r_net_w + (long long)l * D * D, wrnT, D, D, 1);
    convT(out_w + (long long)l * D * D, woutT, D, D, 1);
    convT(ff_w1 + (long long)l * D * FFd, wff1T, D, FFd, 1);
    convT(ff_w2 + (long long)l * FFd * D, wff2T, FFd, D, 1);
    convT(gates_w + (long long)l * 12 * D * D, wgT, D, D, 12);

    unsigned short* xtail = catb + (long long)Mm * Bb * D;  // layer_out bf16

    // q(+r_w_bias), q(+r_r_bias), k, v from transposed qkv_w row slices; r_k from pos_emb
    gemm(GF_TRANSB | GF_OUTBF16, xtail, wqkvT, qwb, r_w_bias,
         T * Bb, D, D, D, D, D, 0, 0, 0, 1, 0.f);
    gemm(GF_TRANSB | GF_OUTBF16, xtail, wqkvT, qrb, r_r_bias,
         T * Bb, D, D, D, D, D, 0, 0, 0, 1, 0.f);
    gemm(GF_TRANSB | GF_OUTBF16, catb, wqkvT + (long long)D * D, kbuf, nullptr,
         KL * Bb, D, D, D, D, D, 0, 0, 0, 1, 0.f);
    gemm(GF_TRANSB | GF_OUTBF16, catb, wqkvT + 2 * (long long)D * D, vbuf, nullptr,
         KL * Bb, D, D, D, D, D, 0, 0, 0, 1, 0.f);
    gemm(GF_TRANSB | GF_OUTBF16, posb, wrnT, rkb, nullptr,
         KL, D, D, D, D, D, 0, 0, 0, 1, 0.f);

    for (int b = 0; b < Bb; ++b) {
      // AC = (q + rw) . K^T   per head (batch z = H)
      gemm(GF_TRANSB, qwb + b * D, kbuf + b * D, scA, nullptr, T, KL, DH,
           Bb * D, Bb * D, KL, DH, DH, (long long)T * KL, H, 0.f);
      // BD = (q + rr) . r_k^T per head
      gemm(GF_TRANSB, qrb + b * D, rkb, scB, nullptr, T, KL, DH,
           Bb * D, H * DH, KL, DH, DH, (long long)T * KL, H, 0.f);
      k_softmax_rel<<<dim3((unsigned)T, (unsigned)H), 256, 0, stream>>>(
          scA, scB, prb, T, KL, Mm, 0.125f);
      // attn_vec = P . V per head, bf16 out into [T][B][H][DH]
      gemm(GF_OUTBF16, prb, vbuf + b * D, avb + b * D, nullptr, T, DH, KL,
           KL, Bb * D, Bb * D, (long long)T * KL, DH, DH, H, 0.f);
    }

    gemm(GF_TRANSB, avb, woutT, attno, nullptr, T * Bb, D, D, D, D, D, 0, 0, 0, 1, 0.f);
    k_layernorm_bf16<<<(unsigned)(T * Bb), 256, 0, stream>>>(attno, yb, D);

    auto grugate = [&](const float* xf, const unsigned short* xb, const unsigned short* yv,
                       const unsigned short* Wg, const float* gbias,
                       float* outf, unsigned short* outb) {
      const long long mm = (long long)D * D;
      gemm(GF_TRANSB, yv, Wg + 0 * mm, t0buf, nullptr, T * Bb, D, D, D, D, D, 0, 0, 0, 1, 0.f);
      gemm(GF_TRANSB | GF_ACC, xb, Wg + 1 * mm, t0buf, nullptr, T * Bb, D, D, D, D, D, 0, 0, 0, 1, 0.f);
      gemm(GF_TRANSB, yv, Wg + 2 * mm, t1buf, nullptr, T * Bb, D, D, D, D, D, 0, 0, 0, 1, 0.f);
      gemm(GF_TRANSB | GF_ACC, xb, Wg + 3 * mm, t1buf, gbias, T * Bb, D, D, D, D, D, 0, 0, 0, 1, -0.1f);
      k_gru_ew1<<<(unsigned)((nTBD + 255) / 256), 256, 0, stream>>>(t0buf, t1buf, xf, rxb, nTBD);
      gemm(GF_TRANSB, yv, Wg + 4 * mm, t2buf, nullptr, T * Bb, D, D, D, D, D, 0, 0, 0, 1, 0.f);
      gemm(GF_TRANSB | GF_ACC, rxb, Wg + 5 * mm, t2buf, nullptr, T * Bb, D, D, D, D, D, 0, 0, 0, 1, 0.f);
      k_gru_ew2<<<(unsigned)((nTBD + 255) / 256), 256, 0, stream>>>(t2buf, t1buf, xf, outf, outb, nTBD);
    };

    const float* xf_l = hs + (long long)l * nTBD;
    grugate(xf_l, xtail, yb, wgT, gates_b + (long long)(l * 2) * D, srcf, srcb);

    gemm(GF_TRANSB | GF_RELU | GF_OUTBF16, srcb, wff1T, h1b, ff_b1 + (long long)l * FFd,
         T * Bb, FFd, D, D, D, FFd, 0, 0, 0, 1, 0.f);
    gemm(GF_TRANSB, h1b, wff2T, h2buf, ff_b2 + (long long)l * D,
         T * Bb, D, FFd, FFd, FFd, D, 0, 0, 0, 1, 0.f);
    k_layernorm_bf16<<<(unsigned)(T * Bb), 256, 0, stream>>>(h2buf, ffbuf, D);

    grugate(srcf, srcb, ffbuf, wgT + 6 * (long long)D * D,
            gates_b + (long long)(l * 2 + 1) * D,
            hs + (long long)(l + 1) * nTBD, xtail);   // bf16 out feeds next layer's cat tail
  }

  long long tot = (long long)(L + 1) * nTBD;
  k_scatter_logits<<<(unsigned)((tot + 255) / 256), 256, 0, stream>>>(hs, logits, T, Bb, D, L + 1);
}
